// TripletGNN_50723563765851
// MI455X (gfx1250) — compile-verified, matrix operand order checked by reading, not run on hardware
//
#include <hip/hip_runtime.h>
#include <hip/hip_bf16.h>

typedef _Float16 v16h __attribute__((ext_vector_type(16)));
typedef float    v8f  __attribute__((ext_vector_type(8)));
typedef float    v2f  __attribute__((ext_vector_type(2)));

#define CCH   256
#define KNB   16
#define TDIM  576
#define QKV3  768
#define NPB   4          // anchor nodes per block (M = 64 rows)
#define NTHR  512        // 16 wave32
#define NTILE 3          // B-column tiles per wave (48 / 16)

union AFrag16 { v16h h; float4 f4[2]; };

// ---------- fp32 -> f16 conversion (weights and x feature table) ----------
__global__ __launch_bounds__(256) void f32_to_f16(const float* __restrict__ src,
                                                  _Float16* __restrict__ dst, long n) {
  long i = (long)blockIdx.x * blockDim.x + threadIdx.x;
  if (i < n) dst[i] = (_Float16)src[i];
}

// ---------- out[:, :256] = x, out[:, 256:] = 0 ----------
__global__ __launch_bounds__(256) void init_out(const float* __restrict__ x,
                                                float* __restrict__ out, int N) {
  long i = (long)blockIdx.x * blockDim.x + threadIdx.x;
  if (i >= (long)N * 512) return;
  int nrow = (int)(i >> 9), c = (int)(i & 511);
  out[i] = (c < CCH) ? x[(long)nrow * CCH + c] : 0.0f;
}

// ---------- main: one block per 4 anchor nodes, 512 threads = 16 wave32 ----------
__global__ __launch_bounds__(NTHR, 1) void triplet_attn(
    const float* __restrict__ x, const _Float16* __restrict__ xh,
    const float* __restrict__ pos,
    const int* __restrict__ anchor, const int* __restrict__ corner,
    const unsigned char* __restrict__ cmask,
    const _Float16* __restrict__ Wh, const float* __restrict__ bias,
    float* __restrict__ out, float* __restrict__ mask_out, int N)
{
  // A-matrix LDS is dead after the GEMM; reuse it for the score partial tiles.
  __shared__ union SU {
    _Float16 A[NPB * KNB][TDIM];            // 73728 B
    float    part[NPB][4][KNB][KNB];        // 16384 B
  } uAS;
  __shared__ float sQKV[NPB * KNB][QKV3];   // 196608 B ; cols 0:256=Q,256:512=K,512:768=V
  __shared__ float sS[NPB][KNB][KNB];
  __shared__ float sBias[QKV3];
  __shared__ int   sCI[NPB][KNB][2];
  __shared__ float sMask[NPB][KNB];
  __shared__ float sCos[NPB][KNB];

  const int tid  = threadIdx.x;
  const int lane = tid & 31;
  const int wave = tid >> 5;          // 0..15
  const int hsel = (lane >> 4) & 1;   // K-half select (ISA 16-bit fragment layout)
  const int cl   = lane & 15;         // col / row-within-tile
  const int n0   = blockIdx.x * NPB;

  for (int i = tid; i < QKV3; i += NTHR) sBias[i] = bias[i];

  // ---- phase 1: geometry per (node, k) on 64 threads ----
  if (tid < NPB * KNB) {
    int nd = tid >> 4, k = tid & 15;
    int nn = n0 + nd;
    if (nn < N) {
      int a = anchor[nn];
      float apx = pos[a * 3 + 0], apy = pos[a * 3 + 1];
      int i0 = corner[(nn * KNB + k) * 2 + 0];
      int i1 = corner[(nn * KNB + k) * 2 + 1];
      float v0x = pos[i0 * 3 + 0] - apx, v0y = pos[i0 * 3 + 1] - apy;
      float v1x = pos[i1 * 3 + 0] - apx, v1y = pos[i1 * 3 + 1] - apy;
      float dot = v0x * v1x + v0y * v1y;
      float nr0 = sqrtf(v0x * v0x + v0y * v0y);
      float nr1 = sqrtf(v1x * v1x + v1y * v1y);
      float cosv = dot / (nr0 * nr1 + 1e-6f);
      float sinz = v0x * v1y - v0y * v1x;
      bool reorder = (sinz < 0.0f);
      if (reorder) { int t = i0; i0 = i1; i1 = t; }
      float new_sinz = reorder ? -sinz : sinz;
      bool m = (cmask[nn * KNB + k] != 0);
      mask_out[nn * KNB + k] = ((new_sinz < -1e-6f) && m) ? 1.0f : 0.0f;
      sCI[nd][k][0] = i0; sCI[nd][k][1] = i1;
      sMask[nd][k] = m ? 1.0f : 0.0f;
      sCos[nd][k]  = cosv;
    } else {
      sCI[nd][k][0] = 0; sCI[nd][k][1] = 0;
      sMask[nd][k] = 0.0f; sCos[nd][k] = 0.0f;
    }
  }
  __syncthreads();

  // ---- phase 2a: sinusoidal embedding -> A[:, 512:576] ----
  for (int p = tid; p < NPB * KNB * 32; p += NTHR) {
    int nd = p >> 9, k = (p >> 5) & 15, i = p & 31;
    float div = __expf(-0.2878231366242557f * (float)i);  // exp(-(2i)*ln(1e4)/64)
    float om  = sCos[nd][k] * div;
    uAS.A[nd * KNB + k][512 + 2 * i]     = (_Float16)__sinf(om);
    uAS.A[nd * KNB + k][512 + 2 * i + 1] = (_Float16)__cosf(om);
  }
  // ---- phase 2b: gather x[ci0]|x[ci1] -> A[:, 0:512] ----
  if (xh) {  // f16 table: straight 16-byte copies, no conversion
    for (int p = tid; p < NPB * KNB * 64; p += NTHR) {
      int row = p >> 6, c8 = (p & 63) * 8;
      int nd = row >> 4, k = row & 15;
      int src = (c8 < CCH) ? sCI[nd][k][0] : sCI[nd][k][1];
      int cc  = c8 & 255;
      *reinterpret_cast<uint4*>(&uAS.A[row][c8]) =
          *reinterpret_cast<const uint4*>(&xh[(long)src * CCH + cc]);
    }
  } else {   // fallback: gather f32 and convert
    for (int p = tid; p < NPB * KNB * 128; p += NTHR) {
      int row = p >> 7, c4 = (p & 127) * 4;
      int nd = row >> 4, k = row & 15;
      int src = (c4 < CCH) ? sCI[nd][k][0] : sCI[nd][k][1];
      int cc  = c4 & 255;
      float4 v = *reinterpret_cast<const float4*>(&x[(long)src * CCH + cc]);
      uAS.A[row][c4 + 0] = (_Float16)v.x;
      uAS.A[row][c4 + 1] = (_Float16)v.y;
      uAS.A[row][c4 + 2] = (_Float16)v.z;
      uAS.A[row][c4 + 3] = (_Float16)v.w;
    }
  }
  __syncthreads();

  // ---- phase 3: QKV GEMM (64x576)@(576x768), v_wmma_f32_16x16x32_f16 ----
  // 48 B-column tiles over 16 waves (3 each); each B fragment feeds 4 M-tiles.
  v8f acc[NTILE][NPB] = {};
  #pragma unroll 1
  for (int kk = 0; kk < TDIM / 32; ++kk) {
    int kb = kk * 32 + hsel * 8;
    AFrag16 af[NPB];
    #pragma unroll
    for (int r = 0; r < NPB; ++r) {
      const _Float16* ap = &uAS.A[r * KNB + cl][kb];
      af[r].f4[0] = *reinterpret_cast<const float4*>(__builtin_assume_aligned(ap, 16));
      af[r].f4[1] = *reinterpret_cast<const float4*>(__builtin_assume_aligned(ap + 16, 16));
    }
    #pragma unroll
    for (int t = 0; t < NTILE; ++t) {
      int gcol = (wave * NTILE + t) * 16 + cl;
      const _Float16* wp = Wh + (size_t)gcol * TDIM + kb;
      AFrag16 bf;
      bf.f4[0] = *reinterpret_cast<const float4*>(wp);
      bf.f4[1] = *reinterpret_cast<const float4*>(wp + 16);
      __builtin_prefetch((const void*)(wp + 32), 0, 3);  // warm next K-slice near WGP
      #pragma unroll
      for (int r = 0; r < NPB; ++r)
        acc[t][r] = __builtin_amdgcn_wmma_f32_16x16x32_f16(
            false, af[r].h, false, bf.h, (short)0, acc[t][r], false, false);
    }
  }
  // branch-free epilogue into the contiguous QKV buffer
  #pragma unroll
  for (int t = 0; t < NTILE; ++t) {
    int g = (wave * NTILE + t) * 16 + cl;
    float b = sBias[g];
    #pragma unroll
    for (int r = 0; r < NPB; ++r) {
      #pragma unroll
      for (int v = 0; v < 8; ++v)
        sQKV[r * KNB + v + hsel * 8][g] = acc[t][r][v] + b;
    }
  }
  __syncthreads();

  // ---- phase 4: scores = Q K^T / 16, fp32 WMMA, 4 waves per node ----
  const int nodeSel = wave >> 2;   // 0..3: which node this wave serves
  const int w4      = wave & 3;    // K-range quarter
  {
    v8f sc = {};
    int base = nodeSel * KNB;
    #pragma unroll 1
    for (int kk = 0; kk < 16; ++kk) {      // 64 channels per wave
      int kb = w4 * 64 + kk * 4 + hsel * 2;
      v2f a2 = *reinterpret_cast<const v2f*>(&sQKV[base + cl][kb]);        // Q
      v2f b2 = *reinterpret_cast<const v2f*>(&sQKV[base + cl][CCH + kb]);  // K^T cols
      sc = __builtin_amdgcn_wmma_f32_16x16x4_f32(
          false, a2, false, b2, (short)0, sc, false, false);
    }
    #pragma unroll
    for (int v = 0; v < 8; ++v) uAS.part[nodeSel][w4][v + hsel * 8][cl] = sc[v];
  }
  __syncthreads();
  for (int p = tid; p < NPB * 256; p += NTHR) {
    int nd = p >> 8, mm = (p >> 4) & 15, cc = p & 15;
    float s = uAS.part[nd][0][mm][cc] + uAS.part[nd][1][mm][cc] +
              uAS.part[nd][2][mm][cc] + uAS.part[nd][3][mm][cc];
    sS[nd][mm][cc] = s * 0.0625f;   // 1/sqrt(256)
  }
  __syncthreads();

  // ---- phase 5: row softmax (64 rows on 64 threads) ----
  if (tid < NPB * KNB) {
    int nd = tid >> 4, r = tid & 15;
    float mx = -3.0e38f;
    #pragma unroll
    for (int j = 0; j < 16; ++j) mx = fmaxf(mx, sS[nd][r][j]);
    float e[16], s = 0.0f;
    #pragma unroll
    for (int j = 0; j < 16; ++j) { e[j] = __expf(sS[nd][r][j] - mx); s += e[j]; }
    float inv = 1.0f / s;
    #pragma unroll
    for (int j = 0; j < 16; ++j) sS[nd][r][j] = e[j] * inv;
  }
  __syncthreads();

  // ---- phase 6: attn @ V, fp32 WMMA, 4 col-tiles per wave; write into Q region ----
  #pragma unroll
  for (int tt = 0; tt < 4; ++tt) {
    int c0 = (w4 * 4 + tt) * 16;
    v8f av = {};
    #pragma unroll
    for (int kk = 0; kk < 4; ++kk) {
      int kb = kk * 4 + hsel * 2;
      v2f a2 = *reinterpret_cast<const v2f*>(&sS[nodeSel][cl][kb]);
      v2f b2;
      b2.x = sQKV[nodeSel * KNB + kb][512 + c0 + cl];
      b2.y = sQKV[nodeSel * KNB + kb + 1][512 + c0 + cl];
      av = __builtin_amdgcn_wmma_f32_16x16x4_f32(
          false, a2, false, b2, (short)0, av, false, false);
    }
    #pragma unroll
    for (int v = 0; v < 8; ++v)
      sQKV[nodeSel * KNB + v + hsel * 8][c0 + cl] = av[v];  // attn_feats (Q dead)
  }
  __syncthreads();

  // ---- phase 7: masked mean over 16 triplets, scatter by anchor ----
  for (int p = tid; p < NPB * CCH; p += NTHR) {
    int nd = p >> 8, c = p & 255;
    int nn = n0 + nd;
    if (nn < N) {
      float denom = 0.0f, s = 0.0f;
      #pragma unroll
      for (int j = 0; j < 16; ++j) {
        denom += sMask[nd][j];
        s += sQKV[nd * KNB + j][c] * sMask[nd][j];
      }
      out[(size_t)anchor[nn] * 512 + CCH + c] = s / denom;
    }
  }
}

extern "C" void kernel_launch(void* const* d_in, const int* in_sizes, int n_in,
                              void* d_out, int out_size, void* d_ws, size_t ws_size,
                              hipStream_t stream) {
  const float* x          = (const float*)d_in[0];
  const float* pos        = (const float*)d_in[1];
  const int*   anchor     = (const int*)d_in[2];
  const int*   corner     = (const int*)d_in[3];
  const unsigned char* cm = (const unsigned char*)d_in[4];
  const float* Ww         = (const float*)d_in[5];
  const float* Wb         = (const float*)d_in[6];
  int N = in_sizes[0] / CCH;

  float* out   = (float*)d_out;
  _Float16* Wh = (_Float16*)d_ws;

  long wn = (long)QKV3 * TDIM;
  long xn = (long)N * CCH;
  f32_to_f16<<<(int)((wn + 255) / 256), 256, 0, stream>>>(Ww, Wh, wn);

  _Float16* xh = nullptr;
  if (ws_size >= (size_t)(wn + xn) * sizeof(_Float16)) {
    xh = Wh + wn;
    f32_to_f16<<<(int)((xn + 255) / 256), 256, 0, stream>>>(x, xh, xn);
  }

  init_out<<<(int)(((long)N * 512 + 255) / 256), 256, 0, stream>>>(x, out, N);

  int nblk = (N + NPB - 1) / NPB;
  triplet_attn<<<nblk, NTHR, 0, stream>>>(x, xh, pos, anchor, corner, cm,
                                          Wh, Wb, out, out + (size_t)N * 512, N);
}
